// VIDEO_MIXER_33741263078046
// MI455X (gfx1250) — matrix operand order repaired and checked by reading, not compiled, not run on hardware
//
#include <hip/hip_runtime.h>
#include <hip/hip_bf16.h>

typedef __attribute__((ext_vector_type(16))) _Float16 v16h;
typedef __attribute__((ext_vector_type(8)))  float    v8f;

#define IMG_DIM 512
#define D_MODEL 768
#define N_LAYER 12
#define D_INNER 1536
#define D_STATE 16
#define D_CONV  4
#define DT_RANK 48
#define BATCH   4
#define SEQ     256
#define NTOK    (BATCH*SEQ)           /* 1024 */
#define XPROJ_N (DT_RANK + 2*D_STATE) /* 80   */

// ---------------------------------------------------------------------------
// WMMA GEMM:  C[M,N] = act(A[M,K] @ B[K,N] + bias) (+ resid)
// act: 0=none, 1=relu, 2=softplus.
// Block tile 64x128, K-step 32, 256 threads = 8 waves (2x4 grid),
// wave -> 32x32 output = 4 x v_wmma_f32_16x16x32_f16 per K-step.
// Double-buffered LDS: next K-tile global loads overlap current WMMAs,
// one barrier per K-step. EDGE uses clamped addresses + select (no exec
// branching). M must be a multiple of 64 (true: M=1024).
// ---------------------------------------------------------------------------
#define BM 64
#define BN 128
#define BK 32
#define LDSS 36   /* u16 row stride: 72 B -> b64-aligned k-quads, odd qword stride (9 x 8 B) */

__device__ __forceinline__ unsigned short f2h_bits(float x) {
  union { _Float16 h; unsigned short u; } cv;
  cv.h = (_Float16)x;
  return cv.u;
}
__device__ __forceinline__ unsigned int pack2h(float a, float b) {
  return (unsigned int)f2h_bits(a) | ((unsigned int)f2h_bits(b) << 16);
}

template <bool EDGE>
__global__ __launch_bounds__(256) void gemm_f16_wmma(
    const float* __restrict__ A, int lda,
    const float* __restrict__ B, int ldb,
    const float* __restrict__ bias,
    const float* __restrict__ resid,
    float* __restrict__ C, int ldc,
    int M, int N, int K, int act)
{
  __shared__ unsigned short sA[2][BM * LDSS];   // [m][k]
  __shared__ unsigned short sB[2][BN * LDSS];   // transposed: [n][k]

  const int tid  = threadIdx.x;
  const int lane = tid & 31;
  const int wid  = tid >> 5;
  const int wm   = wid & 1;      // 2 row slabs of 32
  const int wn   = wid >> 1;     // 4 col slabs of 32
  const int half = lane >> 4;
  const int l16  = lane & 15;

  const int m0 = blockIdx.y * BM;
  const int n0 = blockIdx.x * BN;

  // per-thread staging coordinates
  const int am0 = tid >> 3,          akg0 = tid & 7;          // A group 0
  const int am1 = (tid + 256) >> 3,  akg1 = (tid + 256) & 7;  // A group 1
  int bn_[4], bkg_[4];
  #pragma unroll
  for (int i = 0; i < 4; ++i) { int idx = tid + i*256; bn_[i] = idx & 127; bkg_[i] = idx >> 7; }

  unsigned int au[2][2];   // packed f16 pairs for A (2 groups x 2 dwords)
  unsigned int bu[4][2];   // packed f16 pairs for B (4 groups x 2 dwords)

  auto fetch = [&](int k0) {
    // ---- A tile 64x32 ----
    if (!EDGE) {
      const float4 a0 = *reinterpret_cast<const float4*>(A + (long)(m0 + am0) * lda + k0 + akg0 * 4);
      const float4 a1 = *reinterpret_cast<const float4*>(A + (long)(m0 + am1) * lda + k0 + akg1 * 4);
      au[0][0] = pack2h(a0.x, a0.y); au[0][1] = pack2h(a0.z, a0.w);
      au[1][0] = pack2h(a1.x, a1.y); au[1][1] = pack2h(a1.z, a1.w);
    } else {
      const int ams[2] = {am0, am1}, akgs[2] = {akg0, akg1};
      #pragma unroll
      for (int i = 0; i < 2; ++i) {
        float e[4];
        #pragma unroll
        for (int q = 0; q < 4; ++q) {
          int gk = k0 + akgs[i] * 4 + q;
          int ck = (gk < K) ? gk : (K - 1);            // clamp, no branch
          float v = A[(long)(m0 + ams[i]) * lda + ck];
          e[q] = (gk < K) ? v : 0.0f;                  // select
        }
        au[i][0] = pack2h(e[0], e[1]); au[i][1] = pack2h(e[2], e[3]);
      }
    }
    // ---- B tile 32x128 (stored transposed later) ----
    #pragma unroll
    for (int i = 0; i < 4; ++i) {
      float e[4];
      #pragma unroll
      for (int q = 0; q < 4; ++q) {
        int gk = k0 + bkg_[i] * 4 + q;
        int gn = n0 + bn_[i];
        if (!EDGE) {
          e[q] = B[(long)gk * ldb + gn];
        } else {
          int ck = (gk < K) ? gk : (K - 1);
          int cn = (gn < N) ? gn : (N - 1);
          float v = B[(long)ck * ldb + cn];
          e[q] = (gk < K && gn < N) ? v : 0.0f;
        }
      }
      bu[i][0] = pack2h(e[0], e[1]); bu[i][1] = pack2h(e[2], e[3]);
    }
  };

  auto store_lds = [&](int buf) {
    *reinterpret_cast<uint2*>(&sA[buf][am0 * LDSS + akg0 * 4]) = make_uint2(au[0][0], au[0][1]);
    *reinterpret_cast<uint2*>(&sA[buf][am1 * LDSS + akg1 * 4]) = make_uint2(au[1][0], au[1][1]);
    #pragma unroll
    for (int i = 0; i < 4; ++i)
      *reinterpret_cast<uint2*>(&sB[buf][bn_[i] * LDSS + bkg_[i] * 4]) = make_uint2(bu[i][0], bu[i][1]);
  };

  v8f acc[2][2] = {};

  fetch(0);
  store_lds(0);
  __syncthreads();

  int cur = 0;
  for (int k0 = 0; k0 < K; k0 += BK) {
    const bool has_next = (k0 + BK) < K;
    if (has_next) fetch(k0 + BK);        // global loads overlap WMMAs below

    // ---- fragment gathers (CDNA5 16-bit WMMA VGPR layouts) + WMMA ----
    union Frag { v16h v; unsigned int u[8]; };
    Frag fa[2], fb[2];
    const unsigned short* bufA = sA[cur];
    const unsigned short* bufB = sB[cur];
    #pragma unroll
    for (int mt = 0; mt < 2; ++mt) {
      const unsigned short* pa = bufA + (wm * 32 + mt * 16 + l16) * LDSS;
      #pragma unroll
      for (int r = 0; r < 8; ++r) {
        int k = ((r & 4) ? 16 : 0) + half * 8 + (r & 3) * 2;   // A: 16x32
        fa[mt].u[r] = *(const unsigned int*)(pa + k);
      }
    }
    #pragma unroll
    for (int nt = 0; nt < 2; ++nt) {
      const unsigned short* pb = bufB + (wn * 32 + nt * 16 + l16) * LDSS;
      #pragma unroll
      for (int r = 0; r < 8; ++r) {
        int k = ((r & 4) ? 16 : 0) + half * 8 + (r & 3) * 2;   // B: 32x16
        fb[nt].u[r] = *(const unsigned int*)(pb + k);
      }
    }
    #pragma unroll
    for (int mt = 0; mt < 2; ++mt)
      #pragma unroll
      for (int nt = 0; nt < 2; ++nt)
        acc[mt][nt] = __builtin_amdgcn_wmma_f32_16x16x32_f16(
            false, fa[mt].v, false, fb[nt].v, (short)0, acc[mt][nt],
            false, false);

    if (has_next) {
      // buf cur^1 was last read before the previous barrier -> safe to fill
      store_lds(cur ^ 1);
      __syncthreads();
      cur ^= 1;
    }
  }

  // ---- epilogue: acc VGPR r -> row m = r + half*8, col n = lane%16 ----
  #pragma unroll
  for (int mt = 0; mt < 2; ++mt) {
    #pragma unroll
    for (int nt = 0; nt < 2; ++nt) {
      int gn = n0 + wn * 32 + nt * 16 + l16;
      if (EDGE && gn >= N) continue;
      float bv = bias ? bias[gn] : 0.0f;
      #pragma unroll
      for (int r = 0; r < 8; ++r) {
        int gm = m0 + wm * 32 + mt * 16 + half * 8 + r;
        float x = acc[mt][nt][r] + bv;
        if (act == 1)      x = fmaxf(x, 0.0f);
        else if (act == 2) x = (x > 20.0f) ? x : log1pf(__expf(x));
        if (resid) x += resid[(long)gm * ldc + gn];
        C[(long)gm * ldc + gn] = x;
      }
    }
  }
}

// ---------------------------------------------------------------------------
// Causal depthwise conv (k=4) + bias + SiLU over xz[:, :D_INNER].
// ---------------------------------------------------------------------------
__global__ __launch_bounds__(256) void conv_silu_kernel(
    const float* __restrict__ xz,
    const float* __restrict__ cw,   // [D_INNER][4]
    const float* __restrict__ cb,   // [D_INNER]
    float* __restrict__ out)        // [NTOK][D_INNER]
{
  int idx = blockIdx.x * blockDim.x + threadIdx.x;
  if (idx >= NTOK * D_INNER) return;
  int d   = idx % D_INNER;
  int tok = idx / D_INNER;
  int t   = tok % SEQ;
  float acc = cb[d];
  #pragma unroll
  for (int j = 0; j < D_CONV; ++j) {
    int tt = t - (D_CONV - 1) + j;
    if (tt >= 0)
      acc += cw[d * D_CONV + j] *
             xz[(long)(tok - (D_CONV - 1) + j) * (2 * D_INNER) + d];
  }
  float s = 1.0f / (1.0f + __expf(-acc));
  out[idx] = acc * s;
}

// ---------------------------------------------------------------------------
// Fused selective scan. One thread per (b,d); 16 states in registers.
// B_t/C_t coefficients staged to LDS with GLOBAL_LOAD_ASYNC_TO_LDS_B128
// (ASYNCcnt-tracked CDNA5 async copy, no VGPR round-trip).
// ---------------------------------------------------------------------------
__global__ __launch_bounds__(256) void scan_kernel(
    const float* __restrict__ delta,  // [NTOK][D_INNER] (post softplus)
    const float* __restrict__ xi,     // [NTOK][D_INNER] (post conv+silu)
    const float* __restrict__ xdbl,   // [NTOK][80] : B at +48, C at +64
    const float* __restrict__ xz,     // [NTOK][3072] : res at +D_INNER
    const float* __restrict__ A_log,  // [D_INNER][16] (layer slice)
    const float* __restrict__ Dp,     // [D_INNER]
    float* __restrict__ y)            // [NTOK][D_INNER]
{
  __shared__ float sBC[SEQ * 32];     // per t: [0..15]=B, [16..31]=C  (32 KB)
  const int tid = threadIdx.x;
  const int b   = blockIdx.y;

  // 2048 16-byte quads, 8 per thread, all 16-B aligned (row = 80 f32 = 320 B,
  // +48 f32 = 192 B, j multiple of 4 f32).
  #pragma unroll
  for (int i = 0; i < 8; ++i) {
    int q = tid + i * 256;
    int t = q >> 3, j = (q & 7) * 4;
    unsigned lds_off = (unsigned)(unsigned long long)(&sBC[q * 4]);
    const float* gp = &xdbl[((long)b * SEQ + t) * XPROJ_N + DT_RANK + j];
    asm volatile("global_load_async_to_lds_b128 %0, %1, off"
                 :: "v"(lds_off), "v"((unsigned long long)gp)
                 : "memory");
  }
  asm volatile("s_wait_asynccnt 0x0" ::: "memory");
  __syncthreads();

  int d = blockIdx.x * blockDim.x + tid;
  if (d >= D_INNER) return;

  float a[D_STATE], h[D_STATE];
  #pragma unroll
  for (int n = 0; n < D_STATE; ++n) {
    a[n] = -__expf(A_log[d * D_STATE + n]);
    h[n] = 0.0f;
  }
  const float dpar = Dp[d];

  for (int t = 0; t < SEQ; ++t) {
    long tok = (long)b * SEQ + t;
    float dv = delta[tok * D_INNER + d];
    float xv = xi[tok * D_INNER + d];
    const float* Bp = &sBC[t * 32];
    const float* Cp = Bp + D_STATE;
    float acc = 0.0f;
    #pragma unroll
    for (int n = 0; n < D_STATE; ++n) {
      float dA = __expf(dv * a[n]);
      h[n] = dA * h[n] + dv * Bp[n] * xv;
      acc += h[n] * Cp[n];
    }
    float r = xz[tok * (2 * D_INNER) + D_INNER + d];
    float g = r * (1.0f / (1.0f + __expf(-r)));
    y[tok * D_INNER + d] = (acc + xv * dpar) * g;
  }
}

// ---------------------------------------------------------------------------
// RMSNorm over last dim (D_MODEL=768), one row per block.
// ---------------------------------------------------------------------------
__global__ __launch_bounds__(256) void rmsnorm_kernel(
    const float* __restrict__ x, const float* __restrict__ w,
    float* __restrict__ out)
{
  __shared__ float red[256];
  long row = blockIdx.x;
  const float* xr = x + row * D_MODEL;
  float s = 0.0f;
  for (int i = threadIdx.x; i < D_MODEL; i += 256) { float v = xr[i]; s += v * v; }
  red[threadIdx.x] = s;
  __syncthreads();
  for (int off = 128; off > 0; off >>= 1) {
    if (threadIdx.x < off) red[threadIdx.x] += red[threadIdx.x + off];
    __syncthreads();
  }
  float scale = rsqrtf(red[0] / (float)D_MODEL + 1e-5f);
  for (int i = threadIdx.x; i < D_MODEL; i += 256)
    out[row * D_MODEL + i] = xr[i] * scale * w[i];
}

// ---------------------------------------------------------------------------
extern "C" void kernel_launch(void* const* d_in, const int* in_sizes, int n_in,
                              void* d_out, int out_size, void* d_ws, size_t ws_size,
                              hipStream_t stream)
{
  (void)in_sizes; (void)n_in; (void)out_size; (void)ws_size;

  const float* frames  = (const float*)d_in[0];
  const float* fproj_w = (const float*)d_in[1];
  const float* fproj_b = (const float*)d_in[2];
  const float* norm_w  = (const float*)d_in[3];
  const float* in_w    = (const float*)d_in[4];
  const float* conv_w  = (const float*)d_in[5];
  const float* conv_b  = (const float*)d_in[6];
  const float* xproj_w = (const float*)d_in[7];
  const float* dt_w    = (const float*)d_in[8];
  const float* dt_b    = (const float*)d_in[9];
  const float* A_log   = (const float*)d_in[10];
  const float* D_param = (const float*)d_in[11];
  const float* out_w   = (const float*)d_in[12];
  const float* normf_w = (const float*)d_in[13];

  float* ws = (float*)d_ws;
  float* h     = ws; ws += (long)NTOK * D_MODEL;
  float* xn    = ws; ws += (long)NTOK * D_MODEL;
  float* xz    = ws; ws += (long)NTOK * 2 * D_INNER;
  float* xic   = ws; ws += (long)NTOK * D_INNER;
  float* xdbl  = ws; ws += (long)NTOK * XPROJ_N;
  float* delta = ws; ws += (long)NTOK * D_INNER;
  float* yb    = ws; ws += (long)NTOK * D_INNER;

  auto gemm = [&](const float* Ap, int lda, const float* Bp, int ldb,
                  const float* bias, const float* res, float* Cp, int ldc,
                  int M, int N, int K, int act) {
    dim3 g((N + BN - 1) / BN, M / BM);
    if ((N % BN) || (K % BK))
      gemm_f16_wmma<true><<<g, 256, 0, stream>>>(Ap, lda, Bp, ldb, bias, res,
                                                 Cp, ldc, M, N, K, act);
    else
      gemm_f16_wmma<false><<<g, 256, 0, stream>>>(Ap, lda, Bp, ldb, bias, res,
                                                  Cp, ldc, M, N, K, act);
  };

  // h = relu(frames @ fproj_w + fproj_b)
  gemm(frames, IMG_DIM, fproj_w, D_MODEL, fproj_b, nullptr,
       h, D_MODEL, NTOK, D_MODEL, IMG_DIM, 1);

  for (int l = 0; l < N_LAYER; ++l) {
    rmsnorm_kernel<<<NTOK, 256, 0, stream>>>(h, norm_w + (long)l * D_MODEL, xn);

    // xz = xn @ in_w[l]   (1024 x 768 x 3072)
    gemm(xn, D_MODEL, in_w + (long)l * D_MODEL * 2 * D_INNER, 2 * D_INNER,
         nullptr, nullptr, xz, 2 * D_INNER, NTOK, 2 * D_INNER, D_MODEL, 0);

    // xic = silu(causal_depthwise_conv(xz[:, :D_INNER]) + conv_b)
    conv_silu_kernel<<<(NTOK * D_INNER) / 256, 256, 0, stream>>>(
        xz, conv_w + (long)l * D_INNER * D_CONV, conv_b + (long)l * D_INNER, xic);

    // xdbl = xic @ xproj_w[l]   (1024 x 1536 x 80)  [EDGE: N=80]
    gemm(xic, D_INNER, xproj_w + (long)l * D_INNER * XPROJ_N, XPROJ_N,
         nullptr, nullptr, xdbl, XPROJ_N, NTOK, XPROJ_N, D_INNER, 0);

    // delta = softplus(xdbl[:, :48] @ dt_w[l] + dt_b[l])  [EDGE: K=48]
    gemm(xdbl, XPROJ_N, dt_w + (long)l * DT_RANK * D_INNER, D_INNER,
         dt_b + (long)l * D_INNER, nullptr, delta, D_INNER,
         NTOK, D_INNER, DT_RANK, 2);

    // yb = gated selective scan
    scan_kernel<<<dim3(D_INNER / 256, BATCH), 256, 0, stream>>>(
        delta, xic, xdbl, xz, A_log + (long)l * D_INNER * D_STATE,
        D_param + (long)l * D_INNER, yb);

    // h = h + yb @ out_w[l]   (1024 x 1536 x 768)
    gemm(yb, D_INNER, out_w + (long)l * D_INNER * D_MODEL, D_MODEL,
         nullptr, h, h, D_MODEL, NTOK, D_MODEL, D_INNER, 0);
  }

  rmsnorm_kernel<<<NTOK, 256, 0, stream>>>(h, normf_w, (float*)d_out);
}
